// AttentionBlock_79783312490716
// MI455X (gfx1250) — matrix operand order repaired and checked by reading, not compiled, not run on hardware
//
#include <hip/hip_runtime.h>
#include <hip/hip_bf16.h>

#define B_   2
#define C_   1024
#define T_   2048
#define H_   16
#define CH   64
#define G_   32
#define CPG  32
#define QK_SCALE 0.35355339059327373f  // 64^-0.25

typedef __bf16 bf;
typedef __attribute__((ext_vector_type(2)))  __bf16 v2bf;
typedef __attribute__((ext_vector_type(4)))  __bf16 v4bf;
typedef __attribute__((ext_vector_type(8)))  __bf16 v8bf;
typedef __attribute__((ext_vector_type(16))) __bf16 v16bf;
typedef __attribute__((ext_vector_type(4)))  float  v4f;
typedef __attribute__((ext_vector_type(8)))  float  v8f;
typedef int v4i __attribute__((vector_size(16)));

// ---- CDNA5 async global->LDS path (guarded; falls back to sync copy) ----
#if defined(__has_builtin)
#  if __has_builtin(__builtin_amdgcn_global_load_async_to_lds_b128)
#    define HAVE_ASYNC_LDS 1
#  else
#    define HAVE_ASYNC_LDS 0
#  endif
#  if __has_builtin(__builtin_amdgcn_s_wait_asynccnt)
#    define HAVE_WAIT_ASYNC 1
#  else
#    define HAVE_WAIT_ASYNC 0
#  endif
#else
#  define HAVE_ASYNC_LDS 0
#  define HAVE_WAIT_ASYNC 0
#endif

#if HAVE_ASYNC_LDS
typedef __attribute__((address_space(1))) v4i* gp128;
typedef __attribute__((address_space(3))) v4i* lp128;
#endif

static __device__ __forceinline__ void copy16_lds(const bf* __restrict__ g, bf* l) {
#if HAVE_ASYNC_LDS
    __builtin_amdgcn_global_load_async_to_lds_b128((gp128)g, (lp128)l, 0, 0);
#else
    *(v8bf*)l = *(const v8bf*)g;
#endif
}

static __device__ __forceinline__ void wait_async_lds() {
#if HAVE_ASYNC_LDS
#  if HAVE_WAIT_ASYNC
    __builtin_amdgcn_s_wait_asynccnt(0);
#  else
    asm volatile("s_wait_asynccnt 0x0" ::: "memory");
#  endif
#endif
}

static __device__ __forceinline__ v16bf cat16(v8bf a, v8bf b) {
    return __builtin_shufflevector(a, b, 0,1,2,3,4,5,6,7,8,9,10,11,12,13,14,15);
}

// ---------------- Kernel 1: GroupNorm statistics ------------------------
__global__ void gn_stats(const float* __restrict__ x, float* __restrict__ stats) {
    const int bg = blockIdx.x;                  // b*G_ + g
    const float* xp = x + (size_t)bg * CPG * T_;
    const int N = CPG * T_;
    float s = 0.f, ss = 0.f;
    for (int i = threadIdx.x; i < N; i += 256) {
        float v = xp[i];
        s += v; ss += v * v;
    }
    __shared__ float rs[256], rq[256];
    rs[threadIdx.x] = s; rq[threadIdx.x] = ss;
    __syncthreads();
    for (int st = 128; st > 0; st >>= 1) {
        if ((int)threadIdx.x < st) {
            rs[threadIdx.x] += rs[threadIdx.x + st];
            rq[threadIdx.x] += rq[threadIdx.x + st];
        }
        __syncthreads();
    }
    if (threadIdx.x == 0) {
        float mu  = rs[0] / (float)N;
        float var = rq[0] / (float)N - mu * mu;
        stats[bg * 2 + 0] = mu;
        stats[bg * 2 + 1] = rsqrtf(var + 1e-5f);
    }
}

// ---------------- Kernel 2: normalize + transpose to hT[b][t][c] bf16 ---
__global__ void gn_apply(const float* __restrict__ x, const float* __restrict__ stats,
                         const float* __restrict__ gw, const float* __restrict__ gb,
                         bf* __restrict__ hT) {
    size_t id = (size_t)blockIdx.x * 256 + threadIdx.x;   // over B*T*(C/2)
    int t  = (int)(id % T_);
    int cp = (int)((id / T_) % (C_ / 2));
    int b  = (int)(id / ((size_t)T_ * (C_ / 2)));
    int c  = cp * 2;
    int g  = c / CPG;
    float mu = stats[(b * G_ + g) * 2 + 0];
    float rs = stats[(b * G_ + g) * 2 + 1];
    float v0 = x[((size_t)b * C_ + c) * T_ + t];
    float v1 = x[((size_t)b * C_ + c + 1) * T_ + t];
    float h0 = (v0 - mu) * rs * gw[c]     + gb[c];
    float h1 = (v1 - mu) * rs * gw[c + 1] + gb[c + 1];
    v2bf pk; pk[0] = (bf)h0; pk[1] = (bf)h1;
    *(v2bf*)(hT + ((size_t)b * T_ + t) * C_ + c) = pk;
}

// ---------------- Kernel 3: QKV GEMM (bf16 WMMA, double-buffered) -------
// M=3072, K=1024, N=2048 per batch. Block tile 128x64, K-step 32, 8 waves.
__global__ void qkv_gemm(const float* __restrict__ W, const float* __restrict__ bias,
                         const bf* __restrict__ hT,
                         bf* __restrict__ qT, bf* __restrict__ kT, bf* __restrict__ vv) {
    __shared__ bf As[2][128][32];
    __shared__ bf Bs[2][64][32];
    const int m0 = blockIdx.x * 128;
    const int n0 = blockIdx.y * 64;
    const int b  = blockIdx.z;
    const int tid  = threadIdx.x;
    const int lane = tid & 31;
    const int wv   = tid >> 5;
    v8f acc[4] = {};

    auto loadA = [&](int buf, int k0) {
        #pragma unroll
        for (int p = 0; p < 4; ++p) {
            int r  = (tid >> 3) + 32 * p;
            int cg = (tid & 7) * 4;
            v4f f = *(const v4f*)(W + (size_t)(m0 + r) * C_ + k0 + cg);
            v4bf o4;
            #pragma unroll
            for (int i = 0; i < 4; ++i) o4[i] = (bf)f[i];
            *(v4bf*)&As[buf][r][cg] = o4;
        }
    };
    auto loadB = [&](int buf, int k0) {
        int n  = tid >> 2;
        int cg = (tid & 3) * 8;
        copy16_lds(hT + ((size_t)b * T_ + n0 + n) * C_ + k0 + cg, &Bs[buf][n][cg]);
    };

    loadA(0, 0);
    loadB(0, 0);
    wait_async_lds();
    __syncthreads();

    const int NK = C_ / 32;
    for (int kt = 0; kt < NK; ++kt) {
        const int cur = kt & 1;
        if (kt + 1 < NK) {               // prefetch next tile into other buffer
            loadB(cur ^ 1, (kt + 1) * 32);
            loadA(cur ^ 1, (kt + 1) * 32);
        }
        const int row = wv * 16 + (lane & 15);
        const int alo = (lane < 16) ? 0 : 8;
        v16bf af = cat16(*(v8bf*)&As[cur][row][alo], *(v8bf*)&As[cur][row][alo + 16]);
        const int kb = (lane < 16) ? 0 : 8;
        #pragma unroll
        for (int j = 0; j < 4; ++j) {
            int n = j * 16 + (lane & 15);
            v16bf bfg = cat16(*(v8bf*)&Bs[cur][n][kb], *(v8bf*)&Bs[cur][n][kb + 16]);
            acc[j] = __builtin_amdgcn_wmma_f32_16x16x32_bf16(
                false, af, false, bfg, (short)0, acc[j], false, false);
        }
        wait_async_lds();
        __syncthreads();
    }

    // Epilogue: bias, q/k scale, scatter (o = h*192 + r; r<64 q, <128 k, else v)
    const int o_base = m0 + wv * 16;
    #pragma unroll
    for (int j = 0; j < 4; ++j) {
        #pragma unroll
        for (int r = 0; r < 8; ++r) {
            int o = o_base + r + ((lane >= 16) ? 8 : 0);
            int t = n0 + j * 16 + (lane & 15);
            float v = acc[j][r] + bias[o];
            int h  = o / 192;
            int rr = o % 192;
            int cc = rr & 63;
            int bh = b * H_ + h;
            if (rr < 64) {
                qT[((size_t)bh * T_ + t) * CH + cc] = (bf)(v * QK_SCALE);
            } else if (rr < 128) {
                kT[((size_t)bh * T_ + t) * CH + cc] = (bf)(v * QK_SCALE);
            } else {
                vv[((size_t)bh * CH + cc) * T_ + t] = (bf)v;
            }
        }
    }
}

// ---------------- Kernel 4: flash attention (double-buffered K/V) -------
__global__ void attn(const bf* __restrict__ qT, const bf* __restrict__ kT,
                     const bf* __restrict__ vv, const float* __restrict__ mask,
                     bf* __restrict__ aT) {
    __shared__ bf Qs[64][64];
    __shared__ bf Ks[2][64][64];
    __shared__ bf Vs[2][64][64];
    __shared__ bf Ps[4][16][64];
    const int t0 = blockIdx.x * 64;
    const int bh = blockIdx.y;
    const int b  = bh / H_, h = bh % H_;
    const int tid = threadIdx.x, lane = tid & 31, wv = tid >> 5;  // 128 thr / 4 waves

    auto loadKV = [&](int buf, int s0) {
        #pragma unroll
        for (int p = 0; p < 4; ++p) {
            int idx = p * 128 + tid;
            int r = idx >> 3, cg = (idx & 7) * 8;
            copy16_lds(kT + ((size_t)bh * T_ + s0 + r) * CH + cg, &Ks[buf][r][cg]);
            copy16_lds(vv + ((size_t)bh * CH + r) * T_ + s0 + cg, &Vs[buf][r][cg]);
        }
    };

    #pragma unroll
    for (int p = 0; p < 4; ++p) {
        int idx = p * 128 + tid;
        int r = idx >> 3, cg = (idx & 7) * 8;
        copy16_lds(qT + ((size_t)bh * T_ + t0 + r) * CH + cg, &Qs[r][cg]);
    }
    loadKV(0, 0);
    wait_async_lds();
    __syncthreads();

    v16bf qa[2];
    {
        const int row = wv * 16 + (lane & 15);
        const int alo = (lane < 16) ? 0 : 8;
        qa[0] = cat16(*(v8bf*)&Qs[row][alo],      *(v8bf*)&Qs[row][alo + 16]);
        qa[1] = cat16(*(v8bf*)&Qs[row][32 + alo], *(v8bf*)&Qs[row][32 + alo + 16]);
    }

    v8f oacc[4] = {};
    float Mr[8], Lr[8];
    #pragma unroll
    for (int r = 0; r < 8; ++r) { Mr[r] = -3.0e38f; Lr[r] = 0.f; }
    const float* mrow = mask + (size_t)(bh % B_) * T_;   // replicate tile(mask,(H,1))

    const int NS = T_ / 64;
    for (int si = 0; si < NS; ++si) {
        const int cur = si & 1;
        if (si + 1 < NS) loadKV(cur ^ 1, (si + 1) * 64);  // overlap with compute

        // scores: S = q^T k  (16 x 64 per wave)
        v8f sacc[4] = {};
        #pragma unroll
        for (int ks = 0; ks < 2; ++ks) {
            int kb = ks * 32 + ((lane < 16) ? 0 : 8);
            #pragma unroll
            for (int j = 0; j < 4; ++j) {
                int n = j * 16 + (lane & 15);
                v16bf bfg = cat16(*(v8bf*)&Ks[cur][n][kb], *(v8bf*)&Ks[cur][n][kb + 16]);
                sacc[j] = __builtin_amdgcn_wmma_f32_16x16x32_bf16(
                    false, qa[ks], false, bfg, (short)0, sacc[j], false, false);
            }
        }

        // online softmax (xor 1/2/4/8 reduces along rows in the C-layout)
        float mnew[8], corr[8];
        #pragma unroll
        for (int r = 0; r < 8; ++r) {
            float m = fmaxf(fmaxf(sacc[0][r], sacc[1][r]),
                            fmaxf(sacc[2][r], sacc[3][r]));
            m = fmaxf(m, __shfl_xor(m, 1));
            m = fmaxf(m, __shfl_xor(m, 2));
            m = fmaxf(m, __shfl_xor(m, 4));
            m = fmaxf(m, __shfl_xor(m, 8));
            mnew[r] = fmaxf(Mr[r], m);
            corr[r] = __expf(Mr[r] - mnew[r]);
            Mr[r]   = mnew[r];
        }
        float msk[4];
        #pragma unroll
        for (int j = 0; j < 4; ++j) msk[j] = mrow[si * 64 + j * 16 + (lane & 15)];

        const int prow = ((lane >= 16) ? 8 : 0);
        #pragma unroll
        for (int r = 0; r < 8; ++r) {
            float rowsum = 0.f;
            #pragma unroll
            for (int j = 0; j < 4; ++j) {
                float e = __expf(sacc[j][r] - mnew[r]);
                rowsum += e;  // denominator mask-free (mask applied post-softmax)
                Ps[wv][r + prow][j * 16 + (lane & 15)] = (bf)(e * msk[j]);
            }
            rowsum += __shfl_xor(rowsum, 1);
            rowsum += __shfl_xor(rowsum, 2);
            rowsum += __shfl_xor(rowsum, 4);
            rowsum += __shfl_xor(rowsum, 8);
            Lr[r] = Lr[r] * corr[r] + rowsum;
            #pragma unroll
            for (int j = 0; j < 4; ++j) oacc[j][r] *= corr[r];
        }
        // Ps is wave-private and same-wave LDS ops are in-order: no hw barrier.
        __builtin_amdgcn_wave_barrier();

        // O += P * V   (A from Ps, B from Vs[c][s])
        #pragma unroll
        for (int ks = 0; ks < 2; ++ks) {
            int alo = ks * 32 + ((lane < 16) ? 0 : 8);
            int row = lane & 15;
            v16bf pa = cat16(*(v8bf*)&Ps[wv][row][alo], *(v8bf*)&Ps[wv][row][alo + 16]);
            int kb = ks * 32 + ((lane < 16) ? 0 : 8);
            #pragma unroll
            for (int j = 0; j < 4; ++j) {
                int n = j * 16 + (lane & 15);
                v16bf vb = cat16(*(v8bf*)&Vs[cur][n][kb], *(v8bf*)&Vs[cur][n][kb + 16]);
                oacc[j] = __builtin_amdgcn_wmma_f32_16x16x32_bf16(
                    false, pa, false, vb, (short)0, oacc[j], false, false);
            }
        }
        wait_async_lds();
        __syncthreads();
    }

    // write aT[b][t][h*64+c] bf16 (B-operand layout for proj GEMM)
    #pragma unroll
    for (int j = 0; j < 4; ++j) {
        #pragma unroll
        for (int r = 0; r < 8; ++r) {
            int t = t0 + wv * 16 + r + ((lane >= 16) ? 8 : 0);
            int c = h * CH + j * 16 + (lane & 15);
            aT[((size_t)b * T_ + t) * C_ + c] = (bf)(oacc[j][r] / Lr[r]);
        }
    }
}

// ---------------- Kernel 5: output projection + residual ----------------
__global__ void proj_gemm(const float* __restrict__ W, const float* __restrict__ bias,
                          const bf* __restrict__ aT, const float* __restrict__ x,
                          float* __restrict__ out) {
    __shared__ bf As[2][128][32];
    __shared__ bf Bs[2][64][32];
    const int m0 = blockIdx.x * 128;
    const int n0 = blockIdx.y * 64;
    const int b  = blockIdx.z;
    const int tid  = threadIdx.x;
    const int lane = tid & 31;
    const int wv   = tid >> 5;
    v8f acc[4] = {};

    auto loadA = [&](int buf, int k0) {
        #pragma unroll
        for (int p = 0; p < 4; ++p) {
            int r  = (tid >> 3) + 32 * p;
            int cg = (tid & 7) * 4;
            v4f f = *(const v4f*)(W + (size_t)(m0 + r) * C_ + k0 + cg);
            v4bf o4;
            #pragma unroll
            for (int i = 0; i < 4; ++i) o4[i] = (bf)f[i];
            *(v4bf*)&As[buf][r][cg] = o4;
        }
    };
    auto loadB = [&](int buf, int k0) {
        int n  = tid >> 2;
        int cg = (tid & 3) * 8;
        copy16_lds(aT + ((size_t)b * T_ + n0 + n) * C_ + k0 + cg, &Bs[buf][n][cg]);
    };

    loadA(0, 0);
    loadB(0, 0);
    wait_async_lds();
    __syncthreads();

    const int NK = C_ / 32;
    for (int kt = 0; kt < NK; ++kt) {
        const int cur = kt & 1;
        if (kt + 1 < NK) {
            loadB(cur ^ 1, (kt + 1) * 32);
            loadA(cur ^ 1, (kt + 1) * 32);
        }
        const int row = wv * 16 + (lane & 15);
        const int alo = (lane < 16) ? 0 : 8;
        v16bf af = cat16(*(v8bf*)&As[cur][row][alo], *(v8bf*)&As[cur][row][alo + 16]);
        const int kb = (lane < 16) ? 0 : 8;
        #pragma unroll
        for (int j = 0; j < 4; ++j) {
            int n = j * 16 + (lane & 15);
            v16bf bfg = cat16(*(v8bf*)&Bs[cur][n][kb], *(v8bf*)&Bs[cur][n][kb + 16]);
            acc[j] = __builtin_amdgcn_wmma_f32_16x16x32_bf16(
                false, af, false, bfg, (short)0, acc[j], false, false);
        }
        wait_async_lds();
        __syncthreads();
    }

    const int o_base = m0 + wv * 16;
    #pragma unroll
    for (int j = 0; j < 4; ++j) {
        #pragma unroll
        for (int r = 0; r < 8; ++r) {
            int o = o_base + r + ((lane >= 16) ? 8 : 0);
            int t = n0 + j * 16 + (lane & 15);
            size_t idx = ((size_t)b * C_ + o) * T_ + t;
            out[idx] = x[idx] + acc[j][r] + bias[o];
        }
    }
}

// ---------------- Launch -------------------------------------------------
extern "C" void kernel_launch(void* const* d_in, const int* in_sizes, int n_in,
                              void* d_out, int out_size, void* d_ws, size_t ws_size,
                              hipStream_t stream) {
    (void)in_sizes; (void)n_in; (void)out_size; (void)ws_size;
    const float* x      = (const float*)d_in[0];
    const float* mask   = (const float*)d_in[1];
    const float* qkv_w  = (const float*)d_in[2];
    const float* qkv_b  = (const float*)d_in[3];
    const float* proj_w = (const float*)d_in[4];
    const float* proj_b = (const float*)d_in[5];
    const float* gn_w   = (const float*)d_in[6];
    const float* gn_b   = (const float*)d_in[7];
    float* out = (float*)d_out;

    char* ws = (char*)d_ws;
    float* stats = (float*)ws;                       // B_*G_*2 floats
    bf* hT = (bf*)(ws + 4096);                       // [B][T][C] bf16 (8 MB)
    const size_t NE = (size_t)B_ * T_ * C_;
    bf* qT = hT + NE;                                // [BH][T][64]
    bf* kT = qT + NE;                                // [BH][T][64]
    bf* vv = kT + NE;                                // [BH][64][T]
    bf* aT = hT;                                     // reuse hT for attention output

    gn_stats<<<B_ * G_, 256, 0, stream>>>(x, stats);
    gn_apply<<<(unsigned)(NE / 2 / 256), 256, 0, stream>>>(x, stats, gn_w, gn_b, hT);
    qkv_gemm<<<dim3(3 * C_ / 128, T_ / 64, B_), 256, 0, stream>>>(qkv_w, qkv_b, hT, qT, kT, vv);
    attn<<<dim3(T_ / 64, B_ * H_), 128, 0, stream>>>(qT, kT, vv, mask, aT);
    proj_gemm<<<dim3(C_ / 128, T_ / 64, B_), 256, 0, stream>>>(proj_w, proj_b, aT, x, out);
}
